// Generator_56384330661933
// MI455X (gfx1250) — matrix operand order repaired and checked by reading, not compiled, not run on hardware
//
#include <hip/hip_runtime.h>
#include <math.h>

#define BDIM 1024
#define LLEN 100
#define DIN  32
#define HID  128
#define DOUT 32
#define BT   32          // batch rows per workgroup
#define NTHREADS 256     // 8 wave32

typedef __attribute__((ext_vector_type(16))) __bf16       v16bf;
typedef __attribute__((ext_vector_type(8)))  float        v8f;
typedef __attribute__((ext_vector_type(4)))  unsigned int v4u;

union FragU { v4u q[2]; v16bf bf; };

__device__ __forceinline__ unsigned short f2bf(float x) {
  unsigned int u = __builtin_bit_cast(unsigned int, x);
  u += 0x7FFFu + ((u >> 16) & 1u);            // round-to-nearest-even
  return (unsigned short)(u >> 16);
}

#if __has_builtin(__builtin_amdgcn_tanhf)
__device__ __forceinline__ float fast_tanh(float x) { return __builtin_amdgcn_tanhf(x); }
#else
__device__ __forceinline__ float fast_tanh(float x) { return tanhf(x); }
#endif

// B fragment (32x16 K-major per column): 16 contiguous bf16 per lane.
__device__ __forceinline__ v16bf ld_frag_contig(const unsigned short* p) {
  FragU f;
  f.q[0] = *reinterpret_cast<const v4u*>(p);
  f.q[1] = *reinterpret_cast<const v4u*>(p + 8);
  return f.bf;
}
// A fragment (16x32): 16 contiguous bf16 at p (K=koff..koff+15).
__device__ __forceinline__ v16bf ld_fragA(const unsigned short* p) {
  FragU f;
  f.q[0] = *reinterpret_cast<const v4u*>(p);
  f.q[1] = *reinterpret_cast<const v4u*>(p + 8);
  return f.bf;
}

__device__ __forceinline__ v8f wmma_bf16(v16bf a, v16bf b, v8f c) {
  return __builtin_amdgcn_wmma_f32_16x16x32_bf16(false, a, false, b, (short)0, c,
                                                 false, false);
}

// ---------------- prep: f32 -> bf16 weight conversion -----------------------
__global__ void cvt_f32_bf16(const float* __restrict__ src,
                             unsigned short* __restrict__ dst, int n) {
  int i = blockIdx.x * blockDim.x + threadIdx.x;
  if (i < n) dst[i] = f2bf(src[i]);
}

// ---------------- main fused CDE RK4 kernel ---------------------------------
__global__ __launch_bounds__(NTHREADS)
void cde_rk4_kernel(const float* __restrict__ coeffs,   // [B,L,DIN]
                    const float* __restrict__ times,    // [L]
                    const float* __restrict__ init_w,   // [HID,DIN]
                    const float* __restrict__ init_b,   // [HID]
                    const unsigned short* __restrict__ w0,  // [HID,HID] bf16
                    const float* __restrict__ fb0,
                    const unsigned short* __restrict__ w1,
                    const float* __restrict__ fb1,
                    const unsigned short* __restrict__ w2,
                    const float* __restrict__ fb2,
                    const unsigned short* __restrict__ wf,  // [HID*DIN,HID] bf16
                    const float* __restrict__ fbf,          // [HID*DIN]
                    const float* __restrict__ dec_w,        // [DOUT,HID]
                    const float* __restrict__ dec_b,        // [DOUT]
                    float* __restrict__ out) {              // [B,L,DOUT]
  __shared__ __align__(16) unsigned short actA[BT * HID];   // vf input (bf16)
  __shared__ __align__(16) unsigned short actB[BT * HID];   // ping-pong (bf16)
  __shared__ __align__(16) float zS[BT * HID];              // RK4 state
  __shared__ __align__(16) float dzS[BT * HID];             // vf output k
  __shared__ float dxdtS[BT * DIN];
  __shared__ float biasS[3][HID];
  __shared__ float decbS[DOUT];

  const int tid  = threadIdx.x;
  const int lane = tid & 31;
  const int wv   = tid >> 5;               // 0..7
  const int bb   = blockIdx.x * BT;

  const int n     = lane & 15;             // column within 16-wide tile
  const int half  = lane >> 4;             // 0/1: lane half
  const int koffA = half * 16;             // A-frag K offset (elements)
  const int koffB = half * 16;             // B-frag K offset (elements)

  if (tid < HID) {
    biasS[0][tid] = fb0[tid];
    biasS[1][tid] = fb1[tid];
    biasS[2][tid] = fb2[tid];
  }
  if (tid < DOUT) decbS[tid] = dec_b[tid];

  // ---- z0 = coeffs[:,0,:] @ init_w^T + init_b ----
  for (int i = tid; i < BT * HID; i += NTHREADS) {
    int m = i >> 7, h = i & 127;
    const float* x = &coeffs[(size_t)(bb + m) * LLEN * DIN];
    float acc = init_b[h];
#pragma unroll 8
    for (int d = 0; d < DIN; ++d) acc += x[d] * init_w[h * DIN + d];
    zS[i] = acc;
  }
  __syncthreads();

  // ---- decode t = 0 ----
  for (int i = tid; i < BT * DOUT; i += NTHREADS) {
    int m = i >> 5, o = i & 31;
    float acc = decbS[o];
    const float* zr = &zS[m * HID];
    const float* wr = &dec_w[o * HID];
#pragma unroll 8
    for (int k2 = 0; k2 < HID; ++k2) acc += zr[k2] * wr[k2];
    out[((size_t)(bb + m) * LLEN) * DOUT + o] = acc;
  }

  float zacc[16];

#pragma unroll 1
  for (int t = 0; t < LLEN - 1; ++t) {
    const float dt = times[t + 1] - times[t];
    const float invdt = 1.0f / dt;

    // ---- dxdt tile + act = bf16(z) ----
    for (int i = tid; i < BT * DIN; i += NTHREADS) {
      int m = i >> 5, d = i & 31;
      size_t base = ((size_t)(bb + m) * LLEN + t) * DIN + d;
      dxdtS[i] = (coeffs[base + DIN] - coeffs[base]) * invdt;
    }
#pragma unroll
    for (int q = 0; q < 16; ++q) {
      int i = tid + q * NTHREADS;
      actA[i] = f2bf(zS[i]);
    }
    __syncthreads();

#pragma unroll 1
    for (int s = 0; s < 4; ++s) {
      // ================= vector field evaluation =================
      {
        unsigned short* inb  = actA;
        unsigned short* outb = actB;
        const unsigned short* const Wl[3] = {w0, w1, w2};

        // ---- hidden layers 1..3 : [BT x 128] @ [128 x 128]^T ----
#pragma unroll
        for (int layer = 0; layer < 3; ++layer) {
          const unsigned short* W = Wl[layer];
          const int j = wv * 16 + n;
          const float bj = biasS[layer][j];
          v8f c0 = {}, c1 = {};
#pragma unroll
          for (int kb = 0; kb < 4; ++kb) {
            const int kk = kb * 32;
            v16bf bfr = ld_frag_contig(&W[(size_t)j * HID + kk + koffB]);
            v16bf a0  = ld_fragA(&inb[n * HID + kk + koffA]);
            v16bf a1  = ld_fragA(&inb[(16 + n) * HID + kk + koffA]);
            c0 = wmma_bf16(a0, bfr, c0);
            c1 = wmma_bf16(a1, bfr, c1);
          }
#pragma unroll
          for (int v = 0; v < 8; ++v) {
            int mr = half * 8 + v;
            outb[mr * HID + j]        = f2bf(fast_tanh(c0[v] + bj));
            outb[(16 + mr) * HID + j] = f2bf(fast_tanh(c1[v] + bj));
          }
          __syncthreads();
          unsigned short* tsw = inb; inb = outb; outb = tsw;
        }

        // ---- final layer [BT x 128] @ [128 x 4096]^T fused with
        //      tanh + contraction over DIN -> dz [BT x 128] ----
        v16bf af[4][2];                         // A frags hoisted over h-loop
#pragma unroll
        for (int kb = 0; kb < 4; ++kb) {
          af[kb][0] = ld_fragA(&inb[n * HID + kb * 32 + koffA]);
          af[kb][1] = ld_fragA(&inb[(16 + n) * HID + kb * 32 + koffA]);
        }
        float dxA0[8], dxB0[8], dxA1[8], dxB1[8];
#pragma unroll
        for (int v = 0; v < 8; ++v) {
          int mr = half * 8 + v;
          dxA0[v] = dxdtS[mr * DIN + n];
          dxB0[v] = dxdtS[mr * DIN + 16 + n];
          dxA1[v] = dxdtS[(16 + mr) * DIN + n];
          dxB1[v] = dxdtS[(16 + mr) * DIN + 16 + n];
        }
#pragma unroll 1
        for (int hh = 0; hh < 16; ++hh) {
          const int h = wv * 16 + hh;
          const unsigned short* Wa = &wf[(size_t)(h * 32 + n) * HID];
          const unsigned short* Wb = &wf[(size_t)(h * 32 + 16 + n) * HID];
          v8f cA0 = {}, cB0 = {}, cA1 = {}, cB1 = {};
#pragma unroll
          for (int kb = 0; kb < 4; ++kb) {
            const int kk = kb * 32 + koffB;
            v16bf bA = ld_frag_contig(Wa + kk);
            v16bf bB = ld_frag_contig(Wb + kk);
            cA0 = wmma_bf16(af[kb][0], bA, cA0);
            cA1 = wmma_bf16(af[kb][1], bA, cA1);
            cB0 = wmma_bf16(af[kb][0], bB, cB0);
            cB1 = wmma_bf16(af[kb][1], bB, cB1);
          }
          const float bja = fbf[h * 32 + n];
          const float bjb = fbf[h * 32 + 16 + n];
          float s0[8], s1[8];
#pragma unroll
          for (int v = 0; v < 8; ++v) {
            s0[v] = fast_tanh(cA0[v] + bja) * dxA0[v] +
                    fast_tanh(cB0[v] + bjb) * dxB0[v];
            s1[v] = fast_tanh(cA1[v] + bja) * dxA1[v] +
                    fast_tanh(cB1[v] + bjb) * dxB1[v];
          }
          // reduce over the 16 lanes of each half (the d dimension)
#pragma unroll
          for (int mask = 1; mask < 16; mask <<= 1) {
#pragma unroll
            for (int v = 0; v < 8; ++v) {
              s0[v] += __shfl_xor(s0[v], mask);
              s1[v] += __shfl_xor(s1[v], mask);
            }
          }
          if (n == 0) {                          // lanes 0 and 16, exclusive
#pragma unroll
            for (int v = 0; v < 8; ++v) {
              dzS[(half * 8 + v) * HID + h]        = s0[v];
              dzS[(16 + half * 8 + v) * HID + h]   = s1[v];
            }
          }
        }
      }
      __syncthreads();   // dzS complete

      // ---- RK4 combine (zacc lives in registers) + next-stage input ----
      const float aN = (s < 2) ? 0.5f * dt : dt;
      const float wg = (s == 1 || s == 2) ? 2.0f : 1.0f;
#pragma unroll
      for (int q = 0; q < 16; ++q) {
        int i = tid + q * NTHREADS;
        float k = dzS[i];
        zacc[q] = (s == 0) ? k : zacc[q] + wg * k;
        if (s < 3) actA[i] = f2bf(zS[i] + aN * k);
        else       zS[i]  += (dt * (1.0f / 6.0f)) * zacc[q];
      }
      __syncthreads();
    }

    // ---- decode y[:, t+1, :] = z @ dec_w^T + dec_b ----
    for (int i = tid; i < BT * DOUT; i += NTHREADS) {
      int m = i >> 5, o = i & 31;
      float acc = decbS[o];
      const float* zr = &zS[m * HID];
      const float* wr = &dec_w[o * HID];
#pragma unroll 8
      for (int k2 = 0; k2 < HID; ++k2) acc += zr[k2] * wr[k2];
      out[((size_t)(bb + m) * LLEN + (t + 1)) * DOUT + o] = acc;
    }
  }
}

// ---------------------------------------------------------------------------
extern "C" void kernel_launch(void* const* d_in, const int* in_sizes, int n_in,
                              void* d_out, int out_size, void* d_ws,
                              size_t ws_size, hipStream_t stream) {
  (void)in_sizes; (void)n_in; (void)out_size; (void)ws_size;
  const float* coeffs = (const float*)d_in[0];
  const float* times  = (const float*)d_in[1];
  const float* init_w = (const float*)d_in[2];
  const float* init_b = (const float*)d_in[3];
  const float* fw0    = (const float*)d_in[4];
  const float* fb0    = (const float*)d_in[5];
  const float* fw1    = (const float*)d_in[6];
  const float* fb1    = (const float*)d_in[7];
  const float* fw2    = (const float*)d_in[8];
  const float* fb2    = (const float*)d_in[9];
  const float* fwf    = (const float*)d_in[10];
  const float* fbf    = (const float*)d_in[11];
  const float* dec_w  = (const float*)d_in[12];
  const float* dec_b  = (const float*)d_in[13];
  float* out = (float*)d_out;

  unsigned short* ws16 = (unsigned short*)d_ws;
  unsigned short* w0b = ws16;                 // 128*128
  unsigned short* w1b = ws16 + 16384;
  unsigned short* w2b = ws16 + 32768;
  unsigned short* wfb = ws16 + 49152;         // 4096*128

  cvt_f32_bf16<<<(16384 + 255) / 256, 256, 0, stream>>>(fw0, w0b, 16384);
  cvt_f32_bf16<<<(16384 + 255) / 256, 256, 0, stream>>>(fw1, w1b, 16384);
  cvt_f32_bf16<<<(16384 + 255) / 256, 256, 0, stream>>>(fw2, w2b, 16384);
  cvt_f32_bf16<<<(524288 + 255) / 256, 256, 0, stream>>>(fwf, wfb, 524288);

  cde_rk4_kernel<<<BDIM / BT, NTHREADS, 0, stream>>>(
      coeffs, times, init_w, init_b, w0b, fb0, w1b, fb1, w2b, fb2, wfb, fbf,
      dec_w, dec_b, out);
}
